// MambaSelectiveContextAttention_35656818492109
// MI455X (gfx1250) — matrix operand order repaired and checked by reading, not compiled, not run on hardware
//
#include <hip/hip_runtime.h>
#include <hip/hip_bf16.h>

// Problem constants (match reference)
#define Bb 2
#define Tt 2048
#define Cc 768
#define Hh 12
#define Dd 64
#define WIN 256

typedef __attribute__((ext_vector_type(16))) _Float16 v16h;
typedef __attribute__((ext_vector_type(8)))  _Float16 v8h;
typedef __attribute__((ext_vector_type(8)))  float    v8f;
typedef __attribute__((ext_vector_type(4)))  unsigned int v4u;
typedef __attribute__((ext_vector_type(8)))  int      v8i;
typedef __attribute__((ext_vector_type(4)))  int      v4i;

#define NEG_BIG (-1.0e30f)
#define NEG_TH  (-1.0e29f)
// B tile lands in LDS with 40-half (80 B) row stride: 16B-aligned rows, banks
// spread (20 banks per row step -> 16 rows hit distinct 4-bank groups).
#define LDSB_STRIDE 40

__device__ __forceinline__ float sigmoidf_(float x) { return 1.0f / (1.0f + __expf(-x)); }

__device__ __forceinline__ v16h pack16(v8h lo, v8h hi) {
  return __builtin_shufflevector(lo, hi, 0, 1, 2, 3, 4, 5, 6, 7,
                                 8, 9, 10, 11, 12, 13, 14, 15);
}

// Reductions across the 16 lanes holding one C/D-layout row (xor masks < 16 stay in-half)
__device__ __forceinline__ float rmax16(float v) {
#pragma unroll
  for (int off = 1; off < 16; off <<= 1) v = fmaxf(v, __shfl_xor(v, off, 32));
  return v;
}
__device__ __forceinline__ float rsum16(float v) {
#pragma unroll
  for (int off = 1; off < 16; off <<= 1) v += __shfl_xor(v, off, 32);
  return v;
}

// ---------------------------------------------------------------------------
// TDM: DMA a 64-row x 32-half 2D tile (row stride = rs halves) from global to
// LDS, padding each 16-dword row with 4 dwords -> 40-half LDS row stride.
// D# layout per CDNA5 ISA ch.8 (group0: flags/lds/global addr; group1: dims).
// This toolchain exposes the 6-arg builtin: (g0, g1, g2, g3, g4, cpol);
// groups 2/3/4 are unused for a 2-D tensor and zero-filled.
// ---------------------------------------------------------------------------
#if __has_builtin(__builtin_amdgcn_tensor_load_to_lds)
#define HAVE_TDM 1
__device__ __forceinline__ void tdm_load_64x32(const _Float16* gsrc,
                                               unsigned lds_byte_addr, int rs) {
  const unsigned long long ga = (unsigned long long)(uintptr_t)gsrc;
  v4u g0;
  g0[0] = 1u;                                        // count=1, user mode
  g0[1] = lds_byte_addr;                             // lds_addr
  g0[2] = (unsigned)(ga & 0xFFFFFFFFu);              // global_addr[31:0]
  g0[3] = (unsigned)((ga >> 32) & 0x1FFFFFFu)        // global_addr[56:32]
          | (2u << 30);                              // type=2 ("image")
  v8i g1;
  g1[0] = (1 << 16)      // data_size = 2 bytes
        | (1 << 20)      // pad_enable
        | (3 << 22)      // pad_interval: every 16 dwords (one 32-half row)
        | (3 << 25);     // pad_amount: 4 dwords -> 80 B LDS row stride
  g1[1] = (32 << 16);    // atomic_barrier_addr=0 | tensor_dim0 lo16 = 32
  g1[2] = (64 << 16);    // tensor_dim0 hi16 = 0  | tensor_dim1 lo16 = 64
  g1[3] = (32 << 16);    // tensor_dim1 hi16 = 0  | tile_dim0 = 32
  g1[4] = 64;            // tile_dim1 = 64 | tile_dim2 = 0
  g1[5] = rs;            // tensor_dim0_stride[31:0] (data_size units)
  g1[6] = 0;             // stride hi | tensor_dim1_stride lo
  g1[7] = 0;
  v4i gz4 = {0, 0, 0, 0};             // groups 2/3 unused (2-D tensor)
  v8i gz8 = {0, 0, 0, 0, 0, 0, 0, 0}; // extra descriptor group (unused)
  __builtin_amdgcn_tensor_load_to_lds(g0, g1, gz4, gz4, gz8, 0);
}
#else
#define HAVE_TDM 0
#endif

// ---------------------------------------------------------------------------
// Prep A: f32 -> f16 elementwise (n divisible by 4)
// ---------------------------------------------------------------------------
__global__ __launch_bounds__(256) void cvt_f16_kernel(const float* __restrict__ in,
                                                      _Float16* __restrict__ out, int n) {
  const int idx = (blockIdx.x * 256 + threadIdx.x) * 4;
  if (idx < n) {
    float4 v = *(const float4*)(in + idx);
    _Float16 tmp[4] = {(_Float16)v.x, (_Float16)v.y, (_Float16)v.z, (_Float16)v.w};
    *(uint2*)(out + idx) = *(uint2*)tmp;
  }
}

// ---------------------------------------------------------------------------
// Prep B: transpose + convert: out[n*K + k] = (f16) in[k*N + n]   (K,N mult of 32)
// ---------------------------------------------------------------------------
__global__ __launch_bounds__(256) void cvt_transpose_kernel(const float* __restrict__ in,
                                                            _Float16* __restrict__ out,
                                                            int K, int N) {
  __shared__ float tile[32][33];
  const int tx = threadIdx.x & 31;
  const int ty = threadIdx.x >> 5;
  const int k0 = blockIdx.x * 32;
  const int n0 = blockIdx.y * 32;
#pragma unroll
  for (int r = 0; r < 4; r++)
    tile[ty * 4 + r][tx] = in[(size_t)(k0 + ty * 4 + r) * N + n0 + tx];
  __syncthreads();
#pragma unroll
  for (int r = 0; r < 4; r++)
    out[(size_t)(n0 + ty * 4 + r) * K + k0 + tx] = (_Float16)tile[tx][ty * 4 + r];
}

// ---------------------------------------------------------------------------
// Kernel 1: selection gate  g[b,t] = 0.5 + 0.5*sigmoid( (scale*LN(mamba)) . sel_w + sel_b )
// ---------------------------------------------------------------------------
__global__ __launch_bounds__(256) void sel_kernel(
    const float* __restrict__ mraw, const float* __restrict__ lnw,
    const float* __restrict__ lnb, const float* __restrict__ selw,
    const float* __restrict__ selb, const float* __restrict__ msc,
    float* __restrict__ gbuf) {
  __shared__ float rA[256], rB[256];
  const int row = blockIdx.x;
  const int tid = threadIdx.x;
  const float* x = mraw + (size_t)row * Cc;
  float s = 0.f, s2 = 0.f;
  for (int c = tid; c < Cc; c += 256) { float v = x[c]; s += v; s2 += v * v; }
  rA[tid] = s; rB[tid] = s2;
  __syncthreads();
  for (int off = 128; off > 0; off >>= 1) {
    if (tid < off) { rA[tid] += rA[tid + off]; rB[tid] += rB[tid + off]; }
    __syncthreads();
  }
  const float mu  = rA[0] * (1.0f / Cc);
  const float var = rB[0] * (1.0f / Cc) - mu * mu;
  const float rs  = rsqrtf(var + 1e-5f);
  const float sc  = msc[0];
  float d = 0.f;
  for (int c = tid; c < Cc; c += 256) {
    float v = sc * ((x[c] - mu) * rs * lnw[c] + lnb[c]);
    d += v * selw[c];
  }
  __syncthreads();
  rA[tid] = d;
  __syncthreads();
  for (int off = 128; off > 0; off >>= 1) {
    if (tid < off) rA[tid] += rA[tid + off];
    __syncthreads();
  }
  if (tid == 0) gbuf[row] = 0.5f + 0.5f * sigmoidf_(rA[0] + selb[0]);
}

// ---------------------------------------------------------------------------
// Shared GEMM body: per k-chunk, TDM (or fallback manual copy) stages the
// 64x32 B tile into LDS; each of 8 waves runs 4 WMMAs on its 16-row A slice.
// ---------------------------------------------------------------------------
__device__ __forceinline__ void gemm_tiles(const _Float16* __restrict__ arow,
                                           const _Float16* __restrict__ wtile, int K,
                                           _Float16* ldsB, int tid, int m, int hid,
                                           v8f acc[4]) {
#if !HAVE_TDM
  const int sn = tid >> 2;
  const int sk = (tid & 3) * 8;
#endif
  const int wv = tid >> 5;
  for (int kc = 0; kc < K; kc += 32) {
    __syncthreads();  // all waves done reading previous tile
#if HAVE_TDM
    if (wv == 0) {
      tdm_load_64x32(wtile + kc, (unsigned)(uintptr_t)(void*)ldsB, K);
      if (kc + 32 < K) __builtin_prefetch(arow + kc + 32, 0, 1);  // global_prefetch_b8
      __builtin_amdgcn_s_wait_tensorcnt(0);
    }
#else
    *(v8h*)&ldsB[sn * LDSB_STRIDE + sk] = *(const v8h*)&wtile[(size_t)sn * K + kc + sk];
    if (kc + 32 < K) __builtin_prefetch(arow + kc + 32, 0, 1);
#endif
    v16h a = pack16(*(const v8h*)&arow[kc + hid * 8],
                    *(const v8h*)&arow[kc + 16 + hid * 8]);
    __syncthreads();  // tile visible to all waves
#pragma unroll
    for (int ct = 0; ct < 4; ct++) {
      const _Float16* bp = &ldsB[(ct * 16 + m) * LDSB_STRIDE + hid * 8];
      v16h bm = pack16(*(const v8h*)bp, *(const v8h*)(bp + 16));
      acc[ct] = __builtin_amdgcn_wmma_f32_16x16x32_f16(false, a, false, bm,
                                                       (short)0, acc[ct], false, false);
    }
  }
}

// ---------------------------------------------------------------------------
// Kernel 2: QKV GEMM  Xh(f16 [M,K]) @ Wt(f16 [N,K] pre-transposed) + bias
//   -> q,k f16 [B,H,T,D];  v f16 TRANSPOSED [B,H,D,T]
// ---------------------------------------------------------------------------
__global__ __launch_bounds__(256) void qkv_kernel(
    const _Float16* __restrict__ Xh, const _Float16* __restrict__ Wt,
    const float* __restrict__ bias, _Float16* __restrict__ qb,
    _Float16* __restrict__ kbuf, _Float16* __restrict__ vt) {
  __shared__ _Float16 ldsB[64 * LDSB_STRIDE];
  const int K = Cc;
  const int tid  = threadIdx.x;
  const int lane = tid & 31;
  const int m    = lane & 15;
  const int hid  = lane >> 4;
  const int row0 = blockIdx.x * 128 + (tid >> 5) * 16;
  const int col0 = blockIdx.y * 64;
  const _Float16* arow  = Xh + (size_t)(row0 + m) * K;
  const _Float16* wtile = Wt + (size_t)col0 * K;
  v8f acc[4] = {};
  gemm_tiles(arow, wtile, K, ldsB, tid, m, hid, acc);
#pragma unroll
  for (int ct = 0; ct < 4; ct++) {
    const int n = col0 + ct * 16 + m;
    const int which = n / Cc;
    const int c = n % Cc;
    const int hh = c / Dd, dd = c % Dd;
    const float bv = bias[n];
#pragma unroll
    for (int r = 0; r < 8; r++) {
      const int row = row0 + r + 8 * hid;
      const int bI = row / Tt, tI = row % Tt;
      const _Float16 hv = (_Float16)(acc[ct][r] + bv);
      if (which == 0)      qb  [(((size_t)bI * Hh + hh) * Tt + tI) * Dd + dd] = hv;
      else if (which == 1) kbuf[(((size_t)bI * Hh + hh) * Tt + tI) * Dd + dd] = hv;
      else                 vt  [(((size_t)bI * Hh + hh) * Dd + dd) * Tt + tI] = hv;  // [B,H,D,T]
    }
  }
}

// ---------------------------------------------------------------------------
// Kernel 3: dual-softmax flash attention. One wave per 16-query tile of one
// (b,h); streams keys in blocks of 32. Two online softmaxes (local window +
// gate-scaled global) sharing S tiles and V loads. Output f16 [B,T,C].
// ---------------------------------------------------------------------------
__global__ __launch_bounds__(32) void attn_kernel(
    const _Float16* __restrict__ Q, const _Float16* __restrict__ Kb,
    const _Float16* __restrict__ Vt, const float* __restrict__ g,
    const float* __restrict__ lwp, const float* __restrict__ gwp,
    _Float16* __restrict__ O) {
  __shared__ _Float16 ldsL[16 * 32];
  __shared__ _Float16 ldsG[16 * 32];
  const int lane = threadIdx.x & 31;
  const int m   = lane & 15;
  const int hid = lane >> 4;
  const int nqt = Tt / 16;
  const int qt = blockIdx.x % nqt;
  const int hh = (blockIdx.x / nqt) % Hh;
  const int bI = blockIdx.x / (nqt * Hh);
  const int q0 = qt * 16;
  const size_t base  = ((size_t)bI * Hh + hh) * Tt;   // Q/K row base
  const size_t baseD = ((size_t)bI * Hh + hh) * Dd;   // Vt row base

  const _Float16* qrow = Q + (base + q0 + m) * Dd;
  v16h qa0 = pack16(*(const v8h*)&qrow[hid * 16],      *(const v8h*)&qrow[hid * 16 + 8]);
  v16h qa1 = pack16(*(const v8h*)&qrow[32 + hid * 16], *(const v8h*)&qrow[32 + hid * 16 + 8]);

  v8f Ol[4] = {}, Og[4] = {};
  float mL[8], lL[8], mG[8], lG[8];
#pragma unroll
  for (int r = 0; r < 8; r++) { mL[r] = NEG_BIG; mG[r] = NEG_BIG; lL[r] = 0.f; lG[r] = 0.f; }
  const float scale = 0.125f;  // D^-0.5, D=64

  for (int kb = 0; kb < q0 + 16; kb += 32) {
    const bool doLocal = (kb + 31) >= (q0 - WIN);  // uniform across wave

    v8f s[2];
#pragma unroll
    for (int ct = 0; ct < 2; ct++) {
      const _Float16* krow = Kb + (base + kb + ct * 16 + m) * Dd;  // lane%16 = key (N)
      v16h k0 = pack16(*(const v8h*)&krow[hid * 16],      *(const v8h*)&krow[hid * 16 + 8]);
      v16h k1 = pack16(*(const v8h*)&krow[32 + hid * 16], *(const v8h*)&krow[32 + hid * 16 + 8]);
      v8f z = {};
      s[ct] = __builtin_amdgcn_wmma_f32_16x16x32_f16(false, qa0, false, k0, (short)0, z, false, false);
      s[ct] = __builtin_amdgcn_wmma_f32_16x16x32_f16(false, qa1, false, k1, (short)0, s[ct], false, false);
    }
    const float g0 = g[(size_t)bI * Tt + kb + m];
    const float g1 = g[(size_t)bI * Tt + kb + 16 + m];

    float aL[8], aG[8];
#pragma unroll
    for (int r = 0; r < 8; r++) {
      const int i = q0 + r + 8 * hid;
      const int j0 = kb + m, j1 = kb + 16 + m;
      const float raw0 = s[0][r] * scale, raw1 = s[1][r] * scale;

      // ---- global branch: scores * (0.5 + 0.5*sel[j]), causal mask ----
      float sg0 = (j0 > i) ? NEG_BIG : raw0 * g0;
      float sg1 = (j1 > i) ? NEG_BIG : raw1 * g1;
      float nmG = fmaxf(mG[r], rmax16(fmaxf(sg0, sg1)));
      float alG = __expf(mG[r] - nmG);
      float pg0 = (sg0 < NEG_TH) ? 0.0f : __expf(sg0 - nmG);
      float pg1 = (sg1 < NEG_TH) ? 0.0f : __expf(sg1 - nmG);
      lG[r] = lG[r] * alG + rsum16(pg0 + pg1);
      mG[r] = nmG; aG[r] = alG;
      ldsG[(r + 8 * hid) * 32 + m]      = (_Float16)pg0;
      ldsG[(r + 8 * hid) * 32 + 16 + m] = (_Float16)pg1;

      // ---- local branch: raw scores, causal + window mask ----
      float alL = 1.0f;
      if (doLocal) {
        float sl0 = (j0 > i || j0 < i - WIN) ? NEG_BIG : raw0;
        float sl1 = (j1 > i || j1 < i - WIN) ? NEG_BIG : raw1;
        float nmL = fmaxf(mL[r], rmax16(fmaxf(sl0, sl1)));
        alL = __expf(mL[r] - nmL);
        float pl0 = (sl0 < NEG_TH) ? 0.0f : __expf(sl0 - nmL);
        float pl1 = (sl1 < NEG_TH) ? 0.0f : __expf(sl1 - nmL);
        lL[r] = lL[r] * alL + rsum16(pl0 + pl1);
        mL[r] = nmL;
        ldsL[(r + 8 * hid) * 32 + m]      = (_Float16)pl0;
        ldsL[(r + 8 * hid) * 32 + 16 + m] = (_Float16)pl1;
      }
      aL[r] = alL;
    }
    __syncthreads();  // single-wave workgroup: orders LDS write->read

    const _Float16* pg = &ldsG[m * 32 + hid * 8];
    const _Float16* pl = &ldsL[m * 32 + hid * 8];
    v16h paG = pack16(*(const v8h*)pg, *(const v8h*)(pg + 16));
    v16h paL = pack16(*(const v8h*)pl, *(const v8h*)(pl + 16));
    __syncthreads();

    // O += P @ V   (V read from transposed [B,H,D,T]: contiguous per lane)
#pragma unroll
    for (int nt = 0; nt < 4; nt++) {
      const _Float16* vrow = Vt + (baseD + nt * 16 + m) * Tt + kb + hid * 8;
      v16h vbm = pack16(*(const v8h*)vrow, *(const v8h*)(vrow + 16));
#pragma unroll
      for (int r = 0; r < 8; r++) Og[nt][r] *= aG[r];
      Og[nt] = __builtin_amdgcn_wmma_f32_16x16x32_f16(false, paG, false, vbm,
                                                      (short)0, Og[nt], false, false);
      if (doLocal) {
#pragma unroll
        for (int r = 0; r < 8; r++) Ol[nt][r] *= aL[r];
        Ol[nt] = __builtin_amdgcn_wmma_f32_16x16x32_f16(false, paL, false, vbm,
                                                        (short)0, Ol[nt], false, false);
      }
    }
  }

  float wl = sigmoidf_(lwp[0]), wg = sigmoidf_(gwp[0]);
  const float wsum = wl + wg;
  wl /= wsum; wg /= wsum;
#pragma unroll
  for (int nt = 0; nt < 4; nt++) {
#pragma unroll
    for (int r = 0; r < 8; r++) {
      const int row = q0 + r + 8 * hid;
      const float val = wl * Ol[nt][r] / lL[r] + wg * Og[nt][r] / lG[r];
      O[((size_t)bI * Tt + row) * Cc + hh * Dd + nt * 16 + m] = (_Float16)val;
    }
  }
}

// ---------------------------------------------------------------------------
// Kernel 4: output projection  Oh(f16 [M,768]) @ Wpt(f16 [768,768] transposed)
//   + bias -> f32 d_out.
// ---------------------------------------------------------------------------
__global__ __launch_bounds__(256) void proj_kernel(
    const _Float16* __restrict__ Ah, const _Float16* __restrict__ Wt,
    const float* __restrict__ bias, float* __restrict__ out) {
  __shared__ _Float16 ldsB[64 * LDSB_STRIDE];
  const int K = Cc, N = Cc;
  const int tid  = threadIdx.x;
  const int lane = tid & 31;
  const int m    = lane & 15;
  const int hid  = lane >> 4;
  const int row0 = blockIdx.x * 128 + (tid >> 5) * 16;
  const int col0 = blockIdx.y * 64;
  const _Float16* arow  = Ah + (size_t)(row0 + m) * K;
  const _Float16* wtile = Wt + (size_t)col0 * K;
  v8f acc[4] = {};
  gemm_tiles(arow, wtile, K, ldsB, tid, m, hid, acc);
#pragma unroll
  for (int ct = 0; ct < 4; ct++) {
    const int n = col0 + ct * 16 + m;
    const float bv = bias[n];
#pragma unroll
    for (int r = 0; r < 8; r++) {
      const int row = row0 + r + 8 * hid;
      out[(size_t)row * N + n] = acc[ct][r] + bv;
    }
  }
}

// ---------------------------------------------------------------------------
extern "C" void kernel_launch(void* const* d_in, const int* in_sizes, int n_in,
                              void* d_out, int out_size, void* d_ws, size_t ws_size,
                              hipStream_t stream) {
  (void)in_sizes; (void)n_in; (void)out_size; (void)ws_size;
  const float* x        = (const float*)d_in[0];
  const float* mamba    = (const float*)d_in[1];
  const float* c_attn_w = (const float*)d_in[2];
  const float* c_attn_b = (const float*)d_in[3];
  const float* c_proj_w = (const float*)d_in[4];
  const float* c_proj_b = (const float*)d_in[5];
  const float* ln_w     = (const float*)d_in[6];
  const float* ln_b     = (const float*)d_in[7];
  const float* m_scale  = (const float*)d_in[8];
  const float* sel_w    = (const float*)d_in[9];
  const float* sel_b    = (const float*)d_in[10];
  const float* lw       = (const float*)d_in[11];
  const float* gw       = (const float*)d_in[12];
  float* out = (float*)d_out;

  // workspace layout (all offsets 256B-aligned)
  const size_t XHSZ = (size_t)Bb * Tt * Cc * sizeof(_Float16);        // 6.0 MB
  const size_t WQSZ = (size_t)3 * Cc * Cc * sizeof(_Float16);        // 3.5 MB
  const size_t WPSZ = (size_t)Cc * Cc * sizeof(_Float16);            // 1.2 MB
  const size_t QSZ  = (size_t)Bb * Hh * Tt * Dd * sizeof(_Float16);  // 6.0 MB each
  const size_t GSZ  = (size_t)Bb * Tt * sizeof(float);               // 16 KB
  char* ws = (char*)d_ws;
  _Float16* xh   = (_Float16*)(ws);
  _Float16* wqkt = (_Float16*)(ws + XHSZ);
  _Float16* wpt  = (_Float16*)(ws + XHSZ + WQSZ);
  _Float16* qb   = (_Float16*)(ws + XHSZ + WQSZ + WPSZ);
  _Float16* kb   = (_Float16*)(ws + XHSZ + WQSZ + WPSZ + QSZ);
  _Float16* vt   = (_Float16*)(ws + XHSZ + WQSZ + WPSZ + 2 * QSZ);
  float*    gbuf = (float*)(ws + XHSZ + WQSZ + WPSZ + 3 * QSZ);
  _Float16* oh   = (_Float16*)(ws + XHSZ + WQSZ + WPSZ + 3 * QSZ + GSZ);

  // prep: convert X to f16; transpose+convert both weight matrices
  {
    const int nx = Bb * Tt * Cc;
    cvt_f16_kernel<<<(nx / 4 + 255) / 256, 256, 0, stream>>>(x, xh, nx);
    cvt_transpose_kernel<<<dim3(Cc / 32, (3 * Cc) / 32), 256, 0, stream>>>(c_attn_w, wqkt, Cc, 3 * Cc);
    cvt_transpose_kernel<<<dim3(Cc / 32, Cc / 32), 256, 0, stream>>>(c_proj_w, wpt, Cc, Cc);
  }

  // 1) selection gate per key position
  sel_kernel<<<Bb * Tt, 256, 0, stream>>>(mamba, ln_w, ln_b, sel_w, sel_b, m_scale, gbuf);

  // 2) QKV GEMM: M=4096 (32 row-blocks of 128), N=2304 (36 col-blocks of 64)
  qkv_kernel<<<dim3((Bb * Tt) / 128, (3 * Cc) / 64), 256, 0, stream>>>(
      xh, wqkt, c_attn_b, qb, kb, vt);

  // 3) dual-softmax flash attention: one wave per 16-query tile
  attn_kernel<<<Bb * Hh * (Tt / 16), 32, 0, stream>>>(qb, kb, vt, gbuf, lw, gw, oh);

  // 4) output projection: N=768 (12 col-blocks of 64)
  proj_kernel<<<dim3((Bb * Tt) / 128, Cc / 64), 256, 0, stream>>>(
      oh, wpt, c_proj_b, out);
}
